// EPA_12927851561052
// MI455X (gfx1250) — matrix operand order repaired and checked by reading, not compiled, not run on hardware
//
#include <hip/hip_runtime.h>
#include <stdint.h>

// ---------------- types ----------------
typedef __bf16 v16bf __attribute__((ext_vector_type(16)));
typedef float  v8f   __attribute__((ext_vector_type(8)));
typedef float  f32x4 __attribute__((ext_vector_type(4)));
typedef unsigned int u32x4 __attribute__((ext_vector_type(4)));
typedef unsigned int u32x2 __attribute__((ext_vector_type(2)));
typedef unsigned short u16;

// ---------------- problem constants ----------------
constexpr int  Bb = 4;
constexpr int  Nn = 8192;
constexpr int  Cc = 768;
constexpr int  Hh = 4;
constexpr int  Dd = 192;          // C / H
constexpr int  C3 = 3 * Cc;       // 2304
constexpr int  BH = Bb * Hh;      // 16
constexpr long BN = (long)Bb * Nn; // 32768

// ---------------- TDM (Tensor Data Mover) support ----------------
#if defined(__has_builtin)
#if __has_builtin(__builtin_amdgcn_tensor_load_to_lds) && \
    __has_builtin(__builtin_amdgcn_s_wait_tensorcnt)
#define HAVE_TDM 1
#endif
#endif

#ifdef HAVE_TDM
typedef unsigned int uivec4 __attribute__((ext_vector_type(4)));
typedef int          ivec8  __attribute__((ext_vector_type(8)));
typedef int          ivec4  __attribute__((ext_vector_type(4)));

// Issue a TDM 2D tile load (data_size = 2 bytes) : global -> LDS.
// D# packing per CDNA5 ISA ch.8 (group0: count/lds_addr/global_addr/type,
// group1: data_size, tensor dims, tile dims, stride). 2D tile: groups 2/3 zero.
__device__ __forceinline__ void tdm_load_2d(const void* gaddr, unsigned lds_byte_off,
                                            unsigned tensor_d0, unsigned tensor_d1,
                                            unsigned tile_d0, unsigned tile_d1,
                                            unsigned stride0 /* elements */) {
  unsigned long long ga = (unsigned long long)(size_t)gaddr;
  uivec4 g0;
  g0[0] = 0x1u;                                   // count=1 (valid), user mode
  g0[1] = lds_byte_off;                           // lds_addr (bytes)
  g0[2] = (unsigned)ga;                           // global_addr[31:0]
  g0[3] = (unsigned)(ga >> 32) | (2u << 30);      // global_addr[56:32] | type=2
  ivec8 g1;
  g1[0] = (int)(1u << 16);                                      // data_size=2B
  g1[1] = (int)((tensor_d0 & 0xFFFFu) << 16);                   // tensor_dim0 lo16
  g1[2] = (int)((tensor_d0 >> 16) | ((tensor_d1 & 0xFFFFu) << 16));
  g1[3] = (int)((tensor_d1 >> 16) | ((tile_d0 & 0xFFFFu) << 16)); // tile_dim0
  g1[4] = (int)(tile_d1 & 0xFFFFu);                             // tile_dim1, dim2=0
  g1[5] = (int)stride0;                                         // dim0 stride lo32
  g1[6] = 0;                                                    // stride hi / dim1 stride
  g1[7] = 0;
  ivec4 z4 = {0, 0, 0, 0};
#if defined(__clang_major__) && (__clang_major__ >= 23)
  ivec8 z8 = {0, 0, 0, 0, 0, 0, 0, 0};
  __builtin_amdgcn_tensor_load_to_lds(g0, g1, z4, z4, z8, 0);
#else
  __builtin_amdgcn_tensor_load_to_lds(g0, g1, z4, z4, 0);
#endif
}
#endif  // HAVE_TDM

// ---------------- helpers ----------------
__device__ __forceinline__ u16 f32_to_bf16(float f) {
  union { float f; unsigned u; } x; x.f = f;
  unsigned u = x.u + 0x7FFFu + ((x.u >> 16) & 1u);   // round-to-nearest-even
  return (u16)(u >> 16);
}
__device__ __forceinline__ float bf16_to_f32(u16 v) {
  union { unsigned u; float f; } x; x.u = ((unsigned)v) << 16;
  return x.f;
}
__device__ __forceinline__ v8f zero8() {
  v8f z = {0.f,0.f,0.f,0.f,0.f,0.f,0.f,0.f};
  return z;
}
// A-operand fragment (16x32 bf16, M x K): lane<16 -> row, K 0..15;
// lane>=16 -> same rows, K 16..31. p0 points at [row, k0 + half*16].
__device__ __forceinline__ v16bf load_frag_a(const u16* p0) {
  union { u32x4 u[2]; v16bf v; } x;
  x.u[0] = *(const u32x4*)(p0);
  x.u[1] = *(const u32x4*)(p0 + 8);
  return x.v;
}
// B-operand fragment (32x16 bf16, K x N) loaded from B^T row-major:
// lane<16 -> col, K 0..15 ; lane>=16 -> col, K 16..31. p0 -> [col, k0+half*16].
__device__ __forceinline__ v16bf load_frag_b(const u16* p0) {
  union { u32x4 u[2]; v16bf v; } x;
  x.u[0] = *(const u32x4*)(p0);
  x.u[1] = *(const u32x4*)(p0 + 8);
  return x.v;
}
__device__ __forceinline__ v8f wmma_bf16(v16bf a, v16bf b, v8f c) {
  return __builtin_amdgcn_wmma_f32_16x16x32_bf16(false, a, false, b,
                                                 (short)0, c, false, false);
}

// ---------------- kernel 1: f32 -> bf16 conversion ----------------
__global__ __launch_bounds__(256) void cvt_kernel(const float* __restrict__ in,
                                                  u16* __restrict__ out, long n4) {
  long i = (long)blockIdx.x * blockDim.x + threadIdx.x;
  if (i >= n4) return;
  f32x4 f = *(const f32x4*)(in + i * 4);
  u32x2 o;
  o.x = (unsigned)f32_to_bf16(f.x) | ((unsigned)f32_to_bf16(f.y) << 16);
  o.y = (unsigned)f32_to_bf16(f.z) | ((unsigned)f32_to_bf16(f.w) << 16);
  *(u32x2*)(out + i * 4) = o;
}

// ---------------- kernel 2: QKVV projection GEMM (bf16 WMMA) ----------------
// Y[m, j] = sum_c X[m,c] * W[j,c]  (both K-contiguous row-major: NT GEMM)
// Block tile 128x128, 8 waves (4x2), wave tile 32x64 (2 A-frags x 4 B-frags,
// 8 WMMA per k-step -> 1.5 b128 loads per WMMA).
// Scatter epilogue: q,k -> [BH, D, N] bf16 ; v -> [BH, N, D] bf16 (v^T).
__device__ __forceinline__ void store_qkv_tile(const v8f acc, int m0, int j0, int lane,
                                               u16* __restrict__ qb, u16* __restrict__ kb,
                                               u16* __restrict__ vT) {
  const int col = lane & 15, half = lane >> 4;
  const int j = j0 + col;
  const int three = j / Cc;
  const int rem = j - three * Cc;
  const int h = rem / Dd;
  const int d = rem - h * Dd;
  #pragma unroll
  for (int r = 0; r < 8; ++r) {
    const int m = m0 + r + 8 * half;
    const int b = m >> 13;            // / 8192
    const int n = m & (Nn - 1);
    const u16 val = f32_to_bf16(acc[r]);
    const long bh = (long)b * Hh + h;
    if (three == 0)      qb[(bh * Dd + d) * (long)Nn + n] = val;
    else if (three == 1) kb[(bh * Dd + d) * (long)Nn + n] = val;
    else                 vT[(bh * (long)Nn + n) * Dd + d] = val;
  }
}

__global__ __launch_bounds__(256) void qkvv_gemm(const u16* __restrict__ xb,
                                                 const u16* __restrict__ wb,
                                                 u16* __restrict__ qb,
                                                 u16* __restrict__ kb,
                                                 u16* __restrict__ vT) {
  const int lane = threadIdx.x & 31;
  const int wave = threadIdx.x >> 5;                 // 0..7
  const int wm = wave >> 1, wn = wave & 1;
  const int m_base = blockIdx.x * 128 + wm * 32;     // M tile (2x16)
  const int n_base = blockIdx.y * 128 + wn * 64;     // N tile (4x16)
  const int row = lane & 15, half = lane >> 4;

  const u16* ap0 = xb + (long)(m_base + row) * Cc + half * 16;
  const u16* ap1 = ap0 + 16L * Cc;
  const u16* bp0 = wb + (long)(n_base + row) * Cc + half * 16;
  const u16* bp1 = bp0 + 16L * Cc;
  const u16* bp2 = bp0 + 32L * Cc;
  const u16* bp3 = bp0 + 48L * Cc;

  v8f acc[2][4];
  #pragma unroll
  for (int i = 0; i < 2; ++i)
    #pragma unroll
    for (int jn = 0; jn < 4; ++jn) acc[i][jn] = zero8();

  #pragma unroll 2
  for (int k0 = 0; k0 < Cc; k0 += 32) {
    __builtin_prefetch(ap0 + k0 + 64, 0, 1);
    __builtin_prefetch(ap1 + k0 + 64, 0, 1);
    v16bf a0 = load_frag_a(ap0 + k0);
    v16bf a1 = load_frag_a(ap1 + k0);
    v16bf b0 = load_frag_b(bp0 + k0);
    v16bf b1 = load_frag_b(bp1 + k0);
    v16bf b2 = load_frag_b(bp2 + k0);
    v16bf b3 = load_frag_b(bp3 + k0);
    acc[0][0] = wmma_bf16(a0, b0, acc[0][0]);
    acc[0][1] = wmma_bf16(a0, b1, acc[0][1]);
    acc[0][2] = wmma_bf16(a0, b2, acc[0][2]);
    acc[0][3] = wmma_bf16(a0, b3, acc[0][3]);
    acc[1][0] = wmma_bf16(a1, b0, acc[1][0]);
    acc[1][1] = wmma_bf16(a1, b1, acc[1][1]);
    acc[1][2] = wmma_bf16(a1, b2, acc[1][2]);
    acc[1][3] = wmma_bf16(a1, b3, acc[1][3]);
  }
  #pragma unroll
  for (int i = 0; i < 2; ++i)
    #pragma unroll
    for (int jn = 0; jn < 4; ++jn)
      store_qkv_tile(acc[i][jn], m_base + i * 16, n_base + jn * 16, lane, qb, kb, vT);
}

// ---------------- kernel 3: inverse L2 norms of q,k rows (along N) ----------
__global__ __launch_bounds__(256) void norms_kernel(const u16* __restrict__ qb,
                                                    const u16* __restrict__ kb,
                                                    float* __restrict__ inv_nq,
                                                    float* __restrict__ inv_nk) {
  const long base = (long)blockIdx.x * Nn;   // blockIdx.x = bh*D + d
  float sq = 0.f, sk = 0.f;
  for (int i = threadIdx.x; i < Nn; i += 256) {
    float q = bf16_to_f32(qb[base + i]); sq += q * q;
    float k = bf16_to_f32(kb[base + i]); sk += k * k;
  }
  __shared__ float red0[256], red1[256];
  red0[threadIdx.x] = sq; red1[threadIdx.x] = sk;
  __syncthreads();
  for (int s = 128; s > 0; s >>= 1) {
    if (threadIdx.x < s) {
      red0[threadIdx.x] += red0[threadIdx.x + s];
      red1[threadIdx.x] += red1[threadIdx.x + s];
    }
    __syncthreads();
  }
  if (threadIdx.x == 0) {
    inv_nq[blockIdx.x] = 1.f / fmaxf(sqrtf(red0[0]), 1e-12f);
    inv_nk[blockIdx.x] = 1.f / fmaxf(sqrtf(red1[0]), 1e-12f);
  }
}

// ---------------- kernel 4: scores GEMM (K=8192) + fused scale/softmax ------
// One 1024-thread block per (b,h). 192x192 scores, 12x12 tiles, 32 waves x 5.
// q/k K-chunks staged into LDS by the Tensor Data Mover when available.
constexpr unsigned ATTN_LDS_S   = 0;                       // 192*192*4 = 147456 B
constexpr unsigned ATTN_LDS_QCH = 192 * 192 * 4;           // 192*32*2  = 12288 B
constexpr unsigned ATTN_LDS_KCH = ATTN_LDS_QCH + 192 * 32 * 2;

__global__ __launch_bounds__(1024) void attn_kernel(const u16* __restrict__ qb,
                                                    const u16* __restrict__ kb,
                                                    const float* __restrict__ inv_nq,
                                                    const float* __restrict__ inv_nk,
                                                    const float* __restrict__ temperature,
                                                    u16* __restrict__ attnb) {
  extern __shared__ char smem_raw[];
  float* S   = (float*)(smem_raw + ATTN_LDS_S);
  u16*  qch  = (u16*)(smem_raw + ATTN_LDS_QCH);
  u16*  kch  = (u16*)(smem_raw + ATTN_LDS_KCH);

  const int bh   = blockIdx.x;          // b*H + h
  const int lane = threadIdx.x & 31;
  const int wave = threadIdx.x >> 5;    // 0..31
  const int row  = lane & 15, half = lane >> 4;

  // 5 tiles per wave over a 12x12 (padded to 160) tile grid
  int ti[5], tj[5];
  bool valid[5];
  #pragma unroll
  for (int t = 0; t < 5; ++t) {
    int idx = wave * 5 + t;
    ti[t] = idx / 12; tj[t] = idx % 12;
    valid[t] = (ti[t] < 12);
  }
  v8f acc[5];
  #pragma unroll
  for (int t = 0; t < 5; ++t) acc[t] = zero8();

  const u16* qmat = qb + (long)bh * Dd * Nn;
  const u16* kmat = kb + (long)bh * Dd * Nn;

  for (int k0 = 0; k0 < Nn; k0 += 32) {
#ifdef HAVE_TDM
    // TDM: 192x32-element (2B) tile, row stride 8192 elements, global -> LDS.
    if (wave == 0) tdm_load_2d(qmat + k0, ATTN_LDS_QCH, Nn, Dd, 32, Dd, Nn);
    if (wave == 1) tdm_load_2d(kmat + k0, ATTN_LDS_KCH, Nn, Dd, 32, Dd, Nn);
    if (wave < 2) __builtin_amdgcn_s_wait_tensorcnt(0);
#else
    const unsigned int* qsrc = (const unsigned int*)qmat;
    const unsigned int* ksrc = (const unsigned int*)kmat;
    unsigned int* qdst = (unsigned int*)qch;
    unsigned int* kdst = (unsigned int*)kch;
    for (int t = threadIdx.x; t < 192 * 16; t += 1024) {
      int rowd = t >> 4, kk = t & 15;
      long src = (long)rowd * (Nn / 2) + (k0 / 2) + kk;
      qdst[t] = qsrc[src];
      kdst[t] = ksrc[src];
    }
#endif
    __syncthreads();
    #pragma unroll
    for (int t = 0; t < 5; ++t) {
      if (!valid[t]) continue;
      v16bf a = load_frag_a(qch + (ti[t] * 16 + row) * 32 + half * 16);
      v16bf b = load_frag_b(kch + (tj[t] * 16 + row) * 32 + half * 16);
      acc[t] = wmma_bf16(a, b, acc[t]);
    }
    __syncthreads();
  }

  // scatter accs into the shared score matrix
  #pragma unroll
  for (int t = 0; t < 5; ++t) {
    if (!valid[t]) continue;
    #pragma unroll
    for (int r = 0; r < 8; ++r)
      S[(ti[t] * 16 + r + 8 * half) * Dd + tj[t] * 16 + (lane & 15)] = acc[t][r];
  }
  float* sInvK = (float*)qch;  // reuse staging LDS
  if (threadIdx.x < Dd) sInvK[threadIdx.x] = inv_nk[(long)bh * Dd + threadIdx.x];
  __syncthreads();

  // fused scale + softmax over e, one row per thread (d = 0..191)
  if (threadIdx.x < Dd) {
    const int d = threadIdx.x;
    const float scale = inv_nq[(long)bh * Dd + d] * temperature[bh & (Hh - 1)];
    float* rowp = S + d * Dd;
    float mx = -3.402823466e+38f;
    for (int e = 0; e < Dd; ++e) {
      float v = rowp[e] * scale * sInvK[e];
      rowp[e] = v;
      mx = fmaxf(mx, v);
    }
    float sum = 0.f;
    for (int e = 0; e < Dd; ++e) {
      float ex = __expf(rowp[e] - mx);
      rowp[e] = ex;
      sum += ex;
    }
    const float inv = 1.f / sum;
    u16* out = attnb + ((long)bh * Dd + d) * Dd;
    for (int e = 0; e < Dd; ++e) out[e] = f32_to_bf16(rowp[e] * inv);
  }
}

// ---------------- kernel 5: out = attn @ v, scattered to [B,N,C] f32 --------
// grid.x = N/128 blocks, grid.y = bh. Each wave: one 16-wide n column, all 12 m-tiles.
__global__ __launch_bounds__(256) void out_gemm(const u16* __restrict__ attnb,
                                                const u16* __restrict__ vT,
                                                float* __restrict__ out) {
  extern __shared__ char smem_raw[];
  u16* sattn = (u16*)smem_raw;                     // 192*192*2 = 73728 B at offset 0

  const int bh   = blockIdx.y;
  const int b    = bh >> 2, h = bh & (Hh - 1);
  const int lane = threadIdx.x & 31;
  const int wave = threadIdx.x >> 5;               // 0..7
  const int row  = lane & 15, half = lane >> 4;
  const int n0   = blockIdx.x * 128 + wave * 16;

  // stage the 192x192 bf16 attn matrix into LDS
#ifdef HAVE_TDM
  if (wave == 0) {
    // 1D tile: 36864 contiguous 2B elements.
    tdm_load_2d(attnb + (long)bh * Dd * Dd, 0, Dd * Dd, 1, Dd * Dd, 1, Dd * Dd);
    __builtin_amdgcn_s_wait_tensorcnt(0);
  }
#else
  {
    const unsigned int* src = (const unsigned int*)(attnb + (long)bh * Dd * Dd);
    unsigned int* dst = (unsigned int*)sattn;
    for (int t = threadIdx.x; t < Dd * Dd / 2; t += 256) dst[t] = src[t];
  }
#endif
  __syncthreads();

  v8f acc[12];
  #pragma unroll
  for (int mt = 0; mt < 12; ++mt) acc[mt] = zero8();

  const u16* bbase = vT + ((long)bh * Nn + n0 + row) * Dd + half * 16;
  for (int k0 = 0; k0 < Dd; k0 += 32) {
    v16bf bfrag = load_frag_b(bbase + k0);
    #pragma unroll
    for (int mt = 0; mt < 12; ++mt) {
      v16bf afrag = load_frag_a(sattn + (mt * 16 + row) * Dd + k0 + half * 16);
      acc[mt] = wmma_bf16(afrag, bfrag, acc[mt]);
    }
  }

  // epilogue: out[b, n, h*D + d]
  const int col = lane & 15;
  #pragma unroll
  for (int mt = 0; mt < 12; ++mt) {
    #pragma unroll
    for (int r = 0; r < 8; ++r) {
      const int d = mt * 16 + r + 8 * half;
      const int n = n0 + col;
      out[((long)b * Nn + n) * Cc + h * Dd + d] = acc[mt][r];
    }
  }
}

// ---------------- host launcher ----------------
extern "C" void kernel_launch(void* const* d_in, const int* in_sizes, int n_in,
                              void* d_out, int out_size, void* d_ws, size_t ws_size,
                              hipStream_t stream) {
  const float* x    = (const float*)d_in[0];       // [B,N,C]
  const float* W    = (const float*)d_in[1];       // [3C,C]
  const float* temp = (const float*)d_in[2];       // [H,1,1]
  float* out = (float*)d_out;

  // workspace carving (256B aligned)
  char* p = (char*)d_ws;
  auto carve = [&](size_t bytes) {
    void* r = (void*)p;
    p += (bytes + 255) & ~(size_t)255;
    return r;
  };
  u16*   xb     = (u16*)carve(BN * Cc * sizeof(u16));            // 50.3 MB
  u16*   wb     = (u16*)carve((size_t)C3 * Cc * sizeof(u16));    // 3.5 MB
  u16*   qb     = (u16*)carve((size_t)BH * Dd * Nn * sizeof(u16));
  u16*   kb     = (u16*)carve((size_t)BH * Dd * Nn * sizeof(u16));
  u16*   vT     = (u16*)carve((size_t)BH * Nn * Dd * sizeof(u16));
  float* inv_nq = (float*)carve((size_t)BH * Dd * sizeof(float));
  float* inv_nk = (float*)carve((size_t)BH * Dd * sizeof(float));
  u16*   attnb  = (u16*)carve((size_t)BH * Dd * Dd * sizeof(u16));

  // 1) convert x and W to bf16
  {
    long n4x = BN * Cc / 4;                        // 6291456
    long n4w = (long)C3 * Cc / 4;                  // 442368
    cvt_kernel<<<dim3((unsigned)((n4x + 255) / 256)), 256, 0, stream>>>(x, xb, n4x);
    cvt_kernel<<<dim3((unsigned)((n4w + 255) / 256)), 256, 0, stream>>>(W, wb, n4w);
  }
  // 2) QKVV projection (bf16 WMMA) with q/k/v scatter
  qkvv_gemm<<<dim3(BN / 128, C3 / 128), 256, 0, stream>>>(xb, wb, qb, kb, vT);
  // 3) inverse norms along N
  norms_kernel<<<dim3(BH * Dd), 256, 0, stream>>>(qb, kb, inv_nq, inv_nk);
  // 4) scores + softmax (192*192*4 + 2*192*32*2 bytes of LDS)
  attn_kernel<<<dim3(BH), 1024, 192 * 192 * 4 + 2 * 192 * 32 * 2, stream>>>(
      qb, kb, inv_nq, inv_nk, temp, attnb);
  // 5) attn @ v -> output
  out_gemm<<<dim3(Nn / 128, BH), 256, Dd * Dd * 2, stream>>>(attnb, vT, out);
}
